// MultiHeadAttention_34832184770996
// MI455X (gfx1250) — matrix operand order repaired and checked by reading, not compiled, not run on hardware
//
#include <hip/hip_runtime.h>
#include <hip/hip_bf16.h>

// ---------------------------------------------------------------------------
// MI455X (gfx1250) multi-head attention, f16 WMMA with fp32 accumulation.
// Pipeline: cast -> fused QKV projection -> flash attention (async-LDS
// double-buffered K/V streaming) -> output projection.
// Workspace layout (64 MiB total, all f16):
//   xf16[3][4096][1024]   WT[3][1024][1024]   Wo16[1024][1024]
//   Qh[2][16][2048][64]   Kh[2][16][2048][64] Vt[2][16][64][2048]
//   attn[2][2048][1024]
// ---------------------------------------------------------------------------

typedef __attribute__((ext_vector_type(16))) _Float16 v16h;
typedef __attribute__((ext_vector_type(8)))  _Float16 v8h;
typedef __attribute__((ext_vector_type(8)))  float    v8f;
typedef int v4i __attribute__((vector_size(16)));   // matches builtin param type

static constexpr int NHEAD  = 16;
static constexpr int DMODEL = 1024;
static constexpr int HEAD   = 64;
static constexpr int BATCH  = 2;
static constexpr int SEQ    = 2048;
static constexpr int ROWS   = BATCH * SEQ;   // 4096

// --- WMMA helper -----------------------------------------------------------
__device__ __forceinline__ v8f wmma_f16(v16h a, v16h b, v8f c) {
  return __builtin_amdgcn_wmma_f32_16x16x32_f16(
      /*neg_a=*/false, a, /*neg_b=*/false, b,
      /*c_mod=*/(short)0, c, /*reuse_a=*/false, /*reuse_b=*/false);
}

// A-operand (16x32 f16, M x K): lane holds row M = lane%16.
// element e -> K = 8*(lane/16) + e + (e>=8 ? 8 : 0)   (ISA 7.12.2)
__device__ __forceinline__ v16h load_A16(const _Float16* __restrict__ base,
                                         int ld, int lane) {
  const int r = lane & 15, hf = lane >> 4;
  const _Float16* p = base + (size_t)r * ld + hf * 8;
  union { v16h v; v8h h[2]; } u;
  u.h[0] = *(const v8h*)(p);
  u.h[1] = *(const v8h*)(p + 16);
  return u.v;
}

// B-operand (32x16 f16, K x N) taken from row-major M2[n][k] (ld = row stride):
// lane holds col N = lane%16; element e -> K = 8*(lane/16) + e + (e>=8 ? 8 : 0).
__device__ __forceinline__ v16h load_B16(const _Float16* __restrict__ base,
                                         int ld, int lane) {
  const int r = lane & 15, hf = lane >> 4;
  const _Float16* p = base + (size_t)r * ld + hf * 8;
  union { v16h v; v8h h[2]; } u;
  u.h[0] = *(const v8h*)(p);
  u.h[1] = *(const v8h*)(p + 16);
  return u.v;
}

// --- gfx1250 async global->LDS 16-byte copy (ASYNCcnt-tracked) -------------
__device__ __forceinline__ void async_cp16(const _Float16* g, _Float16* l) {
#if __has_builtin(__builtin_amdgcn_global_load_async_to_lds_b128)
  typedef __attribute__((address_space(1))) v4i gvec;  // global v4i*
  typedef __attribute__((address_space(3))) v4i lvec;  // LDS v4i*
  __builtin_amdgcn_global_load_async_to_lds_b128((gvec*)g, (lvec*)l, 0, 0);
#else
  unsigned loff = (unsigned)(size_t)l;   // LDS byte address (low 32 bits)
  asm volatile("global_load_async_to_lds_b128 %0, %1, off"
               :: "v"(loff), "v"(g) : "memory");
#endif
}

// --- Kernel 1: cast query/key/value fp32 -> f16 ----------------------------
__global__ void cast_inputs_k(const float* __restrict__ q,
                              const float* __restrict__ k,
                              const float* __restrict__ v,
                              _Float16* __restrict__ dst) {
  const int per = ROWS * DMODEL;
  int idx = blockIdx.x * blockDim.x + threadIdx.x;
  if (idx >= 3 * per) return;
  int t = idx / per;
  int e = idx - t * per;
  const float* s = (t == 0) ? q : (t == 1) ? k : v;
  dst[idx] = (_Float16)s[e];
}

// --- Kernel 2: cast + transpose weights ------------------------------------
// WT[t][n][d] = W{q,k,v}[h][d][k] with n = h*64+k ; Wo16[n][d] = Wo[n][d].
__global__ void cast_weights_k(const float* __restrict__ Wq,
                               const float* __restrict__ Wk,
                               const float* __restrict__ Wv,
                               const float* __restrict__ Wo,
                               _Float16* __restrict__ WT,
                               _Float16* __restrict__ Wo16) {
  const int per = DMODEL * DMODEL;
  int idx = blockIdx.x * blockDim.x + threadIdx.x;
  if (idx >= 4 * per) return;
  int t = idx / per;
  int o = idx - t * per;
  if (t == 3) { Wo16[o] = (_Float16)Wo[o]; return; }
  int n = o >> 10;       // h*64 + k
  int d = o & 1023;
  const float* W = (t == 0) ? Wq : (t == 1) ? Wk : Wv;
  WT[idx] = (_Float16)W[((n >> 6) << 16) + (d << 6) + (n & 63)];
}

// --- Kernel 3: fused QKV projection ----------------------------------------
// One wave per 16x64 output tile (= one head's columns). 3*256*16 = 12288
// wave-tiles -> 1536 blocks x 256 threads (8 waves).
__global__ void __launch_bounds__(256)
proj_qkv_k(const _Float16* __restrict__ xf16,
           const _Float16* __restrict__ WT,
           const float* __restrict__ bq, const float* __restrict__ bk,
           const float* __restrict__ bv,
           _Float16* __restrict__ Qh, _Float16* __restrict__ Kh,
           _Float16* __restrict__ Vt) {
  const int lane = threadIdx.x & 31;
  const int wave = blockIdx.x * 8 + (threadIdx.x >> 5);
  const int mat  = wave >> 12;          // 0=Q, 1=K, 2=V
  const int rem  = wave & 4095;
  const int m0   = (rem >> 4) * 16;     // global row 0..4095
  const int h    = rem & 15;            // head == 64-col tile
  const int n0   = h * 64;

  const _Float16* x = xf16 + (size_t)mat * ROWS * DMODEL + (size_t)m0 * DMODEL;
  const _Float16* W = WT   + (size_t)mat * DMODEL * DMODEL;

  v8f acc[4] = {};
  #pragma unroll 4
  for (int kb = 0; kb < DMODEL; kb += 32) {
    if (kb + 32 < DMODEL) __builtin_prefetch(x + kb + 32, 0, 1);
    v16h a = load_A16(x + kb, DMODEL, lane);
    #pragma unroll
    for (int t = 0; t < 4; ++t) {
      v16h b = load_B16(W + (size_t)(n0 + t * 16) * DMODEL + kb, DMODEL, lane);
      acc[t] = wmma_f16(a, b, acc[t]);
    }
  }

  const float* bias = (mat == 0) ? bq : (mat == 1) ? bk : bv;
  const int b  = m0 >> 11;
  const int s0 = m0 & 2047;
  const int r = lane & 15, hf = lane >> 4;
  #pragma unroll
  for (int t = 0; t < 4; ++t) {
    const int kcol = t * 16 + r;
    const float bval = bias[n0 + kcol];
    #pragma unroll
    for (int rr = 0; rr < 8; ++rr) {
      float val = acc[t][rr] + bval;
      int s = s0 + rr + 8 * hf;
      if (mat == 2) {  // V transposed: [b][h][k][s]
        Vt[((size_t)(b * NHEAD + h) * HEAD + kcol) * SEQ + s] = (_Float16)val;
      } else {         // Q/K: [b][h][s][k]
        _Float16* dst = (mat == 0) ? Qh : Kh;
        dst[((size_t)(b * NHEAD + h) * SEQ + s) * HEAD + kcol] = (_Float16)val;
      }
    }
  }
}

// --- Kernel 4: flash attention with async-LDS K/V streaming ----------------
// Grid: 2*16*32 = 1024 blocks x 128 threads (4 waves). All 4 waves share one
// (b,h); each wave owns 16 q-rows. Per 32-key chunk, the block stages
// K(32x64) and Vt(64x32) into LDS with global_load_async_to_lds_b128,
// double-buffered so chunk i+1 streams while chunk i is computed.
// Each thread copies 4x16B per chunk -> 4 async instructions per wave.
__device__ __forceinline__ void stage_chunk(const _Float16* __restrict__ Kb,
                                            const _Float16* __restrict__ Vb,
                                            int kb, _Float16* dK, _Float16* dV,
                                            int tid) {
  #pragma unroll
  for (int i = 0; i < 2; ++i) {
    const int u = tid + i * 128;               // 256 16B units each for K, V
    const int krow = u >> 3, koff = (u & 7) * 8;
    async_cp16(Kb + (size_t)(kb + krow) * HEAD + koff, dK + krow * HEAD + koff);
    const int vrow = u >> 2, voff = (u & 3) * 8;
    async_cp16(Vb + (size_t)vrow * SEQ + kb + voff, dV + vrow * 32 + voff);
  }
}

__global__ void __launch_bounds__(128)
attn_k(const _Float16* __restrict__ Qh, const _Float16* __restrict__ Kh,
       const _Float16* __restrict__ Vt, _Float16* __restrict__ attnOut) {
  __shared__ __align__(32) _Float16 ldsK[2][32][HEAD];  // [buf][key][feat]
  __shared__ __align__(32) _Float16 ldsV[2][HEAD][32];  // [buf][feat][key]
  __shared__ __align__(32) _Float16 ldsP[4][16][32];    // per-wave P staging
  const int tid  = threadIdx.x;
  const int lane = tid & 31;
  const int w    = tid >> 5;
  const int bh   = blockIdx.x >> 5;       // b*16 + h
  const int qblk = blockIdx.x & 31;
  const int b = bh >> 4, h = bh & 15;
  const int q0 = qblk * 64 + w * 16;
  const int r = lane & 15, hf = lane >> 4;

  const _Float16* Qb = Qh + (size_t)bh * SEQ * HEAD;
  const _Float16* Kb = Kh + (size_t)bh * SEQ * HEAD;
  const _Float16* Vb = Vt + (size_t)bh * HEAD * SEQ;

  // Q tile A-operands (head dim split in two K=32 chunks), resident all loop.
  const v16h qa0 = load_A16(Qb + (size_t)q0 * HEAD, HEAD, lane);
  const v16h qa1 = load_A16(Qb + (size_t)q0 * HEAD + 32, HEAD, lane);

  v8f o0 = {}, o1 = {}, o2 = {}, o3 = {};
  float m_run[8], l_run[8];
  #pragma unroll
  for (int i = 0; i < 8; ++i) { m_run[i] = -1e30f; l_run[i] = 0.0f; }
  const float scale = 0.125f;  // 1/sqrt(64)

  // Prologue: start streaming chunk 0 into buffer 0.
  stage_chunk(Kb, Vb, 0, &ldsK[0][0][0], &ldsV[0][0][0], tid);

  for (int it = 0; it < SEQ / 32; ++it) {
    const int cur = it & 1, nxt = cur ^ 1;
    const int kb = it * 32;
    __syncthreads();  // all waves done reading buf[nxt] (chunk it-1) before overwrite
    if (it + 1 < SEQ / 32) {
      stage_chunk(Kb, Vb, kb + 32, &ldsK[nxt][0][0], &ldsV[nxt][0][0], tid);
      // 4 newly issued outstanding; async loads complete in order, so <=4
      // guarantees chunk `it`'s copies have landed in LDS.
      asm volatile("s_wait_asynccnt 0x4" ::: "memory");
    } else {
      asm volatile("s_wait_asynccnt 0x0" ::: "memory");
    }
    __syncthreads();  // chunk `it` resident for every wave

    // S tile: 16 q-rows x 32 keys (two 16-col accumulators), K from LDS.
    v8f s0 = {}, s1 = {};
    {
      v16h k00 = load_B16(&ldsK[cur][0][0],   HEAD, lane);
      v16h k01 = load_B16(&ldsK[cur][0][32],  HEAD, lane);
      v16h k10 = load_B16(&ldsK[cur][16][0],  HEAD, lane);
      v16h k11 = load_B16(&ldsK[cur][16][32], HEAD, lane);
      s0 = wmma_f16(qa0, k00, s0); s0 = wmma_f16(qa1, k01, s0);
      s1 = wmma_f16(qa0, k10, s1); s1 = wmma_f16(qa1, k11, s1);
    }
    // Online softmax per row (row M = rr + 8*hf lives in 16 lanes of a half;
    // xor masks 1,2,4,8 never cross the 16-lane boundary).
    #pragma unroll
    for (int rr = 0; rr < 8; ++rr) {
      float a0 = s0[rr] * scale, a1 = s1[rr] * scale;
      float mx = fmaxf(a0, a1);
      mx = fmaxf(mx, __shfl_xor(mx, 1, 32));
      mx = fmaxf(mx, __shfl_xor(mx, 2, 32));
      mx = fmaxf(mx, __shfl_xor(mx, 4, 32));
      mx = fmaxf(mx, __shfl_xor(mx, 8, 32));
      float mnew  = fmaxf(m_run[rr], mx);
      float alpha = __expf(m_run[rr] - mnew);
      float p0 = __expf(a0 - mnew);
      float p1 = __expf(a1 - mnew);
      float rs = p0 + p1;
      rs += __shfl_xor(rs, 1, 32);
      rs += __shfl_xor(rs, 2, 32);
      rs += __shfl_xor(rs, 4, 32);
      rs += __shfl_xor(rs, 8, 32);
      l_run[rr] = l_run[rr] * alpha + rs;
      m_run[rr] = mnew;
      o0[rr] *= alpha; o1[rr] *= alpha; o2[rr] *= alpha; o3[rr] *= alpha;
      // C-layout -> row-major P staging in per-wave LDS (A-layout transpose)
      ldsP[w][rr + 8 * hf][r]      = (_Float16)p0;
      ldsP[w][rr + 8 * hf][16 + r] = (_Float16)p1;
    }
    asm volatile("s_wait_dscnt 0x0" ::: "memory");  // DS in-order per wave
    v16h pa = load_A16(&ldsP[w][0][0], 32, lane);
    // PV: O(16x64) += P(16x32) @ V(32x64); V B-operands from LDS.
    v16h v0 = load_B16(&ldsV[cur][0][0],  32, lane);
    v16h v1 = load_B16(&ldsV[cur][16][0], 32, lane);
    v16h v2 = load_B16(&ldsV[cur][32][0], 32, lane);
    v16h v3 = load_B16(&ldsV[cur][48][0], 32, lane);
    o0 = wmma_f16(pa, v0, o0);
    o1 = wmma_f16(pa, v1, o1);
    o2 = wmma_f16(pa, v2, o2);
    o3 = wmma_f16(pa, v3, o3);
  }

  #pragma unroll
  for (int rr = 0; rr < 8; ++rr) {
    float inv = 1.0f / l_run[rr];
    int s = q0 + rr + 8 * hf;
    size_t base = ((size_t)b * SEQ + s) * DMODEL + h * HEAD;
    attnOut[base +  0 + r] = (_Float16)(o0[rr] * inv);
    attnOut[base + 16 + r] = (_Float16)(o1[rr] * inv);
    attnOut[base + 32 + r] = (_Float16)(o2[rr] * inv);
    attnOut[base + 48 + r] = (_Float16)(o3[rr] * inv);
  }
}

// --- Kernel 5: output projection  out = attn @ Wo^T + bo (fp32 out) --------
// 256*16 = 4096 wave-tiles -> 512 blocks x 256 threads.
__global__ void __launch_bounds__(256)
out_proj_k(const _Float16* __restrict__ attn, const _Float16* __restrict__ Wo16,
           const float* __restrict__ bo, float* __restrict__ out) {
  const int lane = threadIdx.x & 31;
  const int wave = blockIdx.x * 8 + (threadIdx.x >> 5);
  const int m0 = (wave >> 4) * 16;
  const int n0 = (wave & 15) * 64;
  const _Float16* x = attn + (size_t)m0 * DMODEL;

  v8f acc[4] = {};
  #pragma unroll 4
  for (int kb = 0; kb < DMODEL; kb += 32) {
    if (kb + 32 < DMODEL) __builtin_prefetch(x + kb + 32, 0, 1);
    v16h a = load_A16(x + kb, DMODEL, lane);
    #pragma unroll
    for (int t = 0; t < 4; ++t) {
      v16h b = load_B16(Wo16 + (size_t)(n0 + t * 16) * DMODEL + kb, DMODEL, lane);
      acc[t] = wmma_f16(a, b, acc[t]);
    }
  }
  const int r = lane & 15, hf = lane >> 4;
  #pragma unroll
  for (int t = 0; t < 4; ++t) {
    const int n = n0 + t * 16 + r;
    const float bval = bo[n];
    #pragma unroll
    for (int rr = 0; rr < 8; ++rr) {
      out[(size_t)(m0 + rr + 8 * hf) * DMODEL + n] = acc[t][rr] + bval;
    }
  }
}

// ---------------------------------------------------------------------------
extern "C" void kernel_launch(void* const* d_in, const int* in_sizes, int n_in,
                              void* d_out, int out_size, void* d_ws, size_t ws_size,
                              hipStream_t stream) {
  const float* query  = (const float*)d_in[0];
  const float* key_in = (const float*)d_in[1];
  const float* value  = (const float*)d_in[2];
  const float* Wq = (const float*)d_in[3];
  const float* Wk = (const float*)d_in[4];
  const float* Wv = (const float*)d_in[5];
  const float* bq = (const float*)d_in[6];
  const float* bk = (const float*)d_in[7];
  const float* bv = (const float*)d_in[8];
  const float* Wo = (const float*)d_in[9];
  const float* bo = (const float*)d_in[10];
  float* out = (float*)d_out;

  _Float16* xf16 = (_Float16*)d_ws;                                  // 3*4096*1024
  _Float16* WT   = xf16 + (size_t)3 * ROWS * DMODEL;                 // 3*1024*1024
  _Float16* Wo16 = WT   + (size_t)3 * DMODEL * DMODEL;               // 1024*1024
  _Float16* Qh   = Wo16 + (size_t)DMODEL * DMODEL;                   // 2*16*2048*64
  _Float16* Kh   = Qh   + (size_t)BATCH * NHEAD * SEQ * HEAD;
  _Float16* Vt   = Kh   + (size_t)BATCH * NHEAD * SEQ * HEAD;
  _Float16* attn = Vt   + (size_t)BATCH * NHEAD * HEAD * SEQ;        // ends at 64 MiB

  cast_inputs_k<<<(3 * ROWS * DMODEL + 255) / 256, 256, 0, stream>>>(
      query, key_in, value, xf16);
  cast_weights_k<<<(4 * DMODEL * DMODEL + 255) / 256, 256, 0, stream>>>(
      Wq, Wk, Wv, Wo, WT, Wo16);
  proj_qkv_k<<<1536, 256, 0, stream>>>(xf16, WT, bq, bk, bv, Qh, Kh, Vt);
  attn_k<<<1024, 128, 0, stream>>>(Qh, Kh, Vt, attn);
  out_proj_k<<<512, 256, 0, stream>>>(attn, Wo16, bo, out);
}
